// STGATDecoder_678604833048
// MI455X (gfx1250) — matrix operand order repaired and checked by reading, not compiled, not run on hardware
//
#include <hip/hip_runtime.h>

// STGAT LSTM decoder for MI455X (gfx1250, wave32).
// - Embedding folded into gate weights: one fused K=64 x N=240 f16 matrix
//   (4 x-rows | 56 h-rows | constant-1 bias row | pad), 224 gate cols +
//   4 output-projection cols (pred rides as the 15th N-tile of next step).
// - 99-step recurrence with wave-private state, v_wmma_f32_16x16x32_f16.
// - Teacher X loads software-pipelined one step ahead; x patched directly
//   into the WMMA A operand registers (no LDS round-trip on common path).

typedef _Float16 v16h __attribute__((ext_vector_type(16)));
typedef _Float16 v8h  __attribute__((ext_vector_type(8)));
typedef float    v8f  __attribute__((ext_vector_type(8)));
typedef float    v4f  __attribute__((ext_vector_type(4)));

namespace {
constexpr int kBN      = 256 * 128;   // independent rows
constexpr int kT       = 100;
constexpr int kSteps   = 99;          // T-1 scan steps
constexpr int kKZ      = 64;          // padded K: [x(4) | h(56) | bias=1 | 0,0,0]
constexpr int kNCols   = 240;         // 224 gate cols + 16 pred cols (4 used)
constexpr int kGT      = 14;          // gate N-tiles
constexpr int kWaves   = 4;
constexpr int kRPW     = 16;          // rows per wave
constexpr int kRPG     = kWaves * kRPW; // rows per workgroup = 64
}

#if __has_builtin(__builtin_amdgcn_tanhf)
__device__ __forceinline__ float tanh32(float x) {
  return __builtin_amdgcn_tanhf(x);           // v_tanh_f32
}
__device__ __forceinline__ float sigm(float x) {
  return __builtin_fmaf(tanh32(0.5f * x), 0.5f, 0.5f);
}
#else
__device__ __forceinline__ float sigm(float x) {
  float e = __builtin_amdgcn_exp2f(-1.442695040888963f * x);
  return __builtin_amdgcn_rcpf(1.0f + e);
}
__device__ __forceinline__ float tanh32(float x) {
  return 2.0f * sigm(2.0f * x) - 1.0f;
}
#endif

__device__ __forceinline__ v8f wmma16(v16h a, v16h b, v8f c) {
  // D(16x16,f32) = A(16x32,f16) * B(32x16,f16) + C
  return __builtin_amdgcn_wmma_f32_16x16x32_f16(false, a, false, b,
                                                (short)0, c, false, false);
}

// A operand: 16 rows x 32 K from the z-buffer (row-major, stride kKZ f16).
// lanes 0-15: elems0-15=K[kc*32+0..15]; lanes 16-31: K[kc*32+16..31].
__device__ __forceinline__ v16h loadA(const _Float16* zw, int l, int kc) {
  int row  = l & 15;
  int base = kc * 32 + ((l >> 4) << 4);
  union { v16h v; v8h h[2]; } a;
  a.h[0] = *(const v8h*)(zw + row * kKZ + base);
  a.h[1] = *(const v8h*)(zw + row * kKZ + base + 8);
  return a.v;
}

// B operand: 32 K x 16 cols from weights (column-major-by-K, stride kKZ f16).
// lanes 0-15 hold K[kc*32+0..15] of col 16n+l; lanes 16-31 hold K[kc*32+16..31].
__device__ __forceinline__ v16h loadB(const _Float16* wb, int l, int n, int kc) {
  int col = n * 16 + (l & 15);
  int kst = kc * 32 + ((l >> 4) << 4);
  return *(const v16h*)(wb + col * kKZ + kst);
}

// gate scratch: tile n, lane l stores its v8f at float offset (n*32+l)*8.
// value(row R, col C) lives at ((C>>4)*32 + (C&15) + 16*(R>>3))*8 + (R&7).
__device__ __forceinline__ float gload(const float* gw, int row, int C) {
  int ln = (C & 15) + ((row >> 3) << 4);
  return gw[(((C >> 4) * 32 + ln) << 3) + (row & 7)];
}

__global__ __launch_bounds__(128)
void stgat_lstm_decode(const float* __restrict__ X,
                       const float* __restrict__ hd,
                       const float* __restrict__ noise,
                       const float* __restrict__ W_emb,   // (16,4)
                       const float* __restrict__ b_emb,   // (16,)
                       const float* __restrict__ Wi,      // (4,56,16)
                       const float* __restrict__ bi,      // (4,56)
                       const float* __restrict__ Wh,      // (4,56,56)
                       const float* __restrict__ bh,      // (4,56)
                       const float* __restrict__ W_out,   // (4,56)
                       const float* __restrict__ b_out,   // (4,)
                       const int*   __restrict__ use_steps_p,
                       float* __restrict__ out)
{
  __shared__ __align__(128) _Float16 wbuf[kNCols * kKZ];          // 30 KB
  __shared__ __align__(128) _Float16 zbuf[kWaves][kRPW * kKZ];    //  8 KB
  __shared__ __align__(128) float    pxbuf[kWaves][kRPW * 4];     //  1 KB
  __shared__ __align__(128) float    gbuf[kWaves][kGT * 32 * 8];  // 57 KB

  const int tid = threadIdx.x;
  const int us  = use_steps_p[0];

  // ---- build fused weight matrix Wz (K x 240 cols), column-major-by-K ----
  for (int c = tid; c < kNCols; c += 128) {
    _Float16* wc = wbuf + c * kKZ;
    if (c < 224) {
      int g = c / 56, u = c % 56;
      const float* wiRow = Wi + (g * 56 + u) * 16;
      #pragma unroll
      for (int k = 0; k < 4; ++k) {           // x rows: Wi @ W_emb
        float s = 0.f;
        for (int e = 0; e < 16; ++e) s += wiRow[e] * W_emb[e * 4 + k];
        wc[k] = (_Float16)s;
      }
      const float* whRow = Wh + (g * 56 + u) * 56;
      for (int j = 0; j < 56; ++j) wc[4 + j] = (_Float16)whRow[j];
      float bs = bi[g * 56 + u] + bh[g * 56 + u];
      for (int e = 0; e < 16; ++e) bs += wiRow[e] * b_emb[e];
      wc[60] = (_Float16)bs;                  // bias via constant-1 K-lane
      wc[61] = wc[62] = wc[63] = (_Float16)0.f;
    } else {
      for (int q = 0; q < kKZ; ++q) wc[q] = (_Float16)0.f;
      int k = c - 224;
      if (k < 4) {                            // pred cols: h . W_out + b_out
        for (int j = 0; j < 56; ++j) wc[4 + j] = (_Float16)W_out[k * 56 + j];
        wc[60] = (_Float16)b_out[k];
      }
    }
  }

  const int w = tid >> 5, l = tid & 31;
  const int rowBase = blockIdx.x * kRPG + w * kRPW;
  _Float16* zw  = zbuf[w];
  float*    pxw = pxbuf[w];
  float*    gw  = gbuf[w];

  // ---- init z: x=0, h0=[hd|noise], bias lane=1 ----
  for (int idx = l; idx < kRPW * kKZ; idx += 32) {
    int row = idx >> 6, k = idx & 63;
    int gRow = rowBase + row;
    float v = 0.f;
    if (k >= 4 && k < 60) {
      int j = k - 4;
      v = (j < 40) ? hd[gRow * 40 + j] : noise[gRow * 16 + (j - 40)];
    } else if (k == 60) v = 1.f;
    zw[idx] = (_Float16)v;
  }
  // out[:, :, 99, :] = X_last (straight copy)
  if (l < 16) {
    int gRow = rowBase + l;
    v4f xl = *(const v4f*)(X + (gRow * kT + 99) * 4);
    *(v4f*)(out + (gRow * kT + 99) * 4) = xl;
  }
  __syncthreads();   // weights visible; all later state is wave-private

  float cst[28];
  #pragma unroll
  for (int j = 0; j < 28; ++j) cst[j] = 0.f;

  const int kcol = l & 15;
  const int hi   = l >> 4;
  const int rowX = rowBase + (l & 15);     // teacher-load row for this lane

  // software-pipelined teacher x: value for the CURRENT step's teacher write
  v4f xt_cur = *(const v4f*)(X + (rowX * kT + kSteps) * 4);   // X[:, :, 99, :]

  for (int t = 0; t <= kSteps; ++t) {
    // --- A operands (h = h_{t-1} from LDS; x slots stale, patched below) ---
    v16h a0 = loadA(zw, l, 0);
    v16h a1 = loadA(zw, l, 1);

    // --- pred tile (N-tile 14): x rows carry zero weights, stale x is OK ---
    v8f accP = {};
    accP = wmma16(a0, loadB(wbuf, l, 14, 0), accP);
    accP = wmma16(a1, loadB(wbuf, l, 14, 1), accP);

    const bool self = (t > us) && (t > 0) && (t < kSteps);

    // --- extract X_hat_{t-1} = X_{t-1} - pred, write out time 99-t ---
    if (t > 0) {
      if (kcol < 4) {
        float xh[8];
        #pragma unroll
        for (int r = 0; r < 8; ++r) {
          int row = r + (hi << 3);
          xh[r] = pxw[row * 4 + kcol] - accP[r];
          out[((rowBase + row) * kT + (kSteps - t)) * 4 + kcol] = xh[r];
        }
        if (self) {                       // self-feed x_t through LDS
          #pragma unroll
          for (int r = 0; r < 8; ++r) {
            int row = r + (hi << 3);
            zw[row * kKZ + kcol] = (_Float16)xh[r];
            pxw[row * 4 + kcol]  = xh[r];
          }
        }
      }
      if (self) a0 = loadA(zw, l, 0);     // uniform branch: refresh x slots
    }
    if (t == kSteps) break;

    // --- teacher-forced x_t: patch A registers directly (no LDS trip) ---
    if (t == 0 || t <= us) {
      if (l < 16) {
        *(v4f*)(pxw + l * 4) = xt_cur;
        a0[0] = (_Float16)xt_cur[0];
        a0[1] = (_Float16)xt_cur[1];
        a0[2] = (_Float16)xt_cur[2];
        a0[3] = (_Float16)xt_cur[3];
      }
    }
    // issue next step's teacher load now; consumed next iteration
    xt_cur = *(const v4f*)(X + (rowX * kT + (kSteps - t - 1)) * 4);

    // --- gate GEMM: (16 x 64) @ (64 x 224) over 14 N-tiles ---
    #pragma unroll
    for (int n = 0; n < kGT; ++n) {
      v8f acc = {};
      acc = wmma16(a0, loadB(wbuf, l, n, 0), acc);
      acc = wmma16(a1, loadB(wbuf, l, n, 1), acc);
      *(v8f*)(gw + ((n * 32 + l) << 3)) = acc;   // 2x ds_store_b128
    }

    // --- LSTM elementwise: lane owns (row = l&15, units 28*hi + 0..27) ---
    {
      int row = l & 15;
      #pragma unroll
      for (int j = 0; j < 28; ++j) {
        int u = hi * 28 + j;
        float gi = gload(gw, row,       u);
        float gf = gload(gw, row,  56 + u);
        float gg = gload(gw, row, 112 + u);
        float go = gload(gw, row, 168 + u);
        float it = sigm(gi), ft = sigm(gf), ot = sigm(go);
        float gt = tanh32(gg);
        float cn = ft * cst[j] + it * gt;
        cst[j] = cn;
        float hn = ot * tanh32(cn);
        zw[row * kKZ + 4 + u] = (_Float16)hn;   // feeds next step's A operand
      }
    }
  }
}

extern "C" void kernel_launch(void* const* d_in, const int* in_sizes, int n_in,
                              void* d_out, int out_size, void* d_ws, size_t ws_size,
                              hipStream_t stream) {
  const float* X     = (const float*)d_in[0];
  const float* hd    = (const float*)d_in[1];
  const float* noise = (const float*)d_in[2];
  const float* W_emb = (const float*)d_in[3];
  const float* b_emb = (const float*)d_in[4];
  const float* Wi    = (const float*)d_in[5];
  const float* bi    = (const float*)d_in[6];
  const float* Wh    = (const float*)d_in[7];
  const float* bh    = (const float*)d_in[8];
  const float* W_out = (const float*)d_in[9];
  const float* b_out = (const float*)d_in[10];
  const int*   us    = (const int*)d_in[11];
  float* out = (float*)d_out;

  dim3 grid(kBN / kRPG);   // 512 workgroups x 64 rows
  dim3 block(128);         // 4 wave32 waves
  stgat_lstm_decode<<<grid, block, 0, stream>>>(X, hd, noise, W_emb, b_emb,
                                                Wi, bi, Wh, bh, W_out, b_out,
                                                us, out);
}
